// Attention_89902255440825
// MI455X (gfx1250) — matrix operand order repaired and checked by reading, not compiled, not run on hardware
//
#include <hip/hip_runtime.h>
#include <hip/hip_bf16.h>
#include <cstdint>
#include <cstddef>

// ---------------- problem constants (B=1) ----------------
constexpr int SEQ = 2048;
constexpr int HID = 2048;
constexpr int NHQ = 16;
constexpr int NKV = 4;
constexpr int HDM = 128;          // head dim

typedef __attribute__((ext_vector_type(16))) __bf16 v16bf;
typedef __attribute__((ext_vector_type(8)))  __bf16 v8bf;
typedef __attribute__((ext_vector_type(8)))  float  v8f;

// ---------------- WMMA helpers ----------------
__device__ __forceinline__ v8f wmma_bf16(v16bf a, v16bf b, v8f c) {
  return __builtin_amdgcn_wmma_f32_16x16x32_bf16(false, a, false, b,
                                                 (short)0, c, false, false);
}

// A fragment (16x32 bf16, M x K): lane m = lane&15, half = lane>>4
//   elems 0..15 <- K = kk + half*16 + (0..15)  (32B contiguous)
__device__ __forceinline__ v16bf load_a_frag(const __bf16* base) {
  v16bf a;
  *(v8bf*)&a         = *(const v8bf*)(base);
  *(((v8bf*)&a) + 1) = *(const v8bf*)(base + 8);
  return a;
}

// B fragment (32x16 bf16, K x N): lane n = lane&15, half = lane>>4
//   elems 0..15 <- K = kk + half*16 + (0..15)  (32B contiguous)
__device__ __forceinline__ v16bf load_b_frag(const __bf16* base) {
  v16bf b;
  *(v8bf*)&b         = *(const v8bf*)(base);
  *(((v8bf*)&b) + 1) = *(const v8bf*)(base + 8);
  return b;
}

// ---------------- gfx1250 async global->LDS (ASYNCcnt) ----------------
__device__ __forceinline__ unsigned lds_off(const void* p) {
  return (unsigned)(uintptr_t)(__attribute__((address_space(3))) const void*)p;
}
// 16B per lane, global -> LDS, tracked by ASYNCcnt
__device__ __forceinline__ void async_ld_b128(const __bf16* g, unsigned ldsByteOff) {
  asm volatile("global_load_async_to_lds_b128 %0, %1, off"
               :: "v"(ldsByteOff), "v"(g) : "memory");
}
__device__ __forceinline__ void wait_async0() {
  asm volatile("s_wait_asynccnt 0x0" ::: "memory");
}

// ---------------- conversion kernels ----------------
__global__ __launch_bounds__(256) void k_f32_to_bf16(const float* __restrict__ x,
                                                     __bf16* __restrict__ y, int n) {
  int i = blockIdx.x * 256 + threadIdx.x;
  if (i < n) y[i] = (__bf16)x[i];
}

// W is K x N (row-major f32); WT is N x K (row-major bf16)
__global__ __launch_bounds__(256) void k_transpose_bf16(const float* __restrict__ W,
                                                        __bf16* __restrict__ WT,
                                                        int K, int N) {
  long idx = (long)blockIdx.x * 256 + threadIdx.x;
  long total = (long)K * N;
  if (idx >= total) return;
  int n = (int)(idx / K);
  int k = (int)(idx % K);
  WT[idx] = (__bf16)W[(size_t)k * N + n];
}

// ---------------- bf16 WMMA GEMM with async-LDS double-buffered staging ----------------
// block = 256 threads (8 waves) -> 128x256 C tile; wave tile 64x64 (16 WMMA C-tiles).
// Per 32-k slice: A tile 128x32 and B tile 256x32 staged in LDS via
// GLOBAL_LOAD_ASYNC_TO_LDS_B128, double buffered; fragments read with ds_load_b128.
__global__ __launch_bounds__(256)
void k_gemm_bf16(const __bf16* __restrict__ A, const __bf16* __restrict__ BT,
                 const float* __restrict__ bias, float* __restrict__ C,
                 int M, int N, int K) {
  __shared__ __bf16 Ash[2][128 * 32];   // [buf][row*32 + k]
  __shared__ __bf16 Bsh[2][256 * 32];   // [buf][col*32 + k]

  const int tid  = threadIdx.x;
  const int lane = tid & 31;
  const int wave = tid >> 5;
  const int mblk = blockIdx.y * 128;
  const int nblk = blockIdx.x * 256;
  const int lm0  = (wave >> 2) * 64;            // wave's local row base
  const int ln0  = (wave & 3) * 64;             // wave's local col base
  const int lm = lane & 15, half = lane >> 4;

  // cooperative staging map:
  //   B: thread t owns column t      -> 4 x 16B chunks (64B row of k)
  //   A: thread t owns row t>>1      -> 2 x 16B chunks (half a 64B row)
  const __bf16* bcol = BT + (size_t)(nblk + tid) * K;
  const int arow  = tid >> 1;
  const int apart = (tid & 1) * 2;              // chunks {0,1} or {2,3}
  const __bf16* arowp = A + (size_t)(mblk + arow) * K;

  // prologue: stage k-slice 0 into buffer 0
#pragma unroll
  for (int p = 0; p < 4; ++p)
    async_ld_b128(bcol + p * 8, lds_off(&Bsh[0][tid * 32 + p * 8]));
#pragma unroll
  for (int p = 0; p < 2; ++p)
    async_ld_b128(arowp + (apart + p) * 8,
                  lds_off(&Ash[0][arow * 32 + (apart + p) * 8]));

  v8f c[4][4];
#pragma unroll
  for (int i = 0; i < 4; ++i)
#pragma unroll
    for (int j = 0; j < 4; ++j) c[i][j] = (v8f)(0.f);

  for (int kk = 0; kk < K; kk += 32) {
    const int buf = (kk >> 5) & 1;
    wait_async0();        // my stripes of buf are in LDS
    __syncthreads();      // everyone's stripes visible; prev buf fully consumed

    if (kk + 32 < K) {    // prefetch next slice into the other buffer
#pragma unroll
      for (int p = 0; p < 4; ++p)
        async_ld_b128(bcol + kk + 32 + p * 8,
                      lds_off(&Bsh[buf ^ 1][tid * 32 + p * 8]));
#pragma unroll
      for (int p = 0; p < 2; ++p)
        async_ld_b128(arowp + kk + 32 + (apart + p) * 8,
                      lds_off(&Ash[buf ^ 1][arow * 32 + (apart + p) * 8]));
    }

    v16bf a[4], b[4];
#pragma unroll
    for (int i = 0; i < 4; ++i)
      a[i] = load_a_frag(&Ash[buf][(lm0 + i * 16 + lm) * 32 + half * 16]);
#pragma unroll
    for (int j = 0; j < 4; ++j)
      b[j] = load_b_frag(&Bsh[buf][(ln0 + j * 16 + lm) * 32 + half * 16]);
#pragma unroll
    for (int i = 0; i < 4; ++i)
#pragma unroll
      for (int j = 0; j < 4; ++j) c[i][j] = wmma_bf16(a[i], b[j], c[i][j]);
  }

#pragma unroll
  for (int i = 0; i < 4; ++i)
#pragma unroll
    for (int j = 0; j < 4; ++j) {
      const int col = nblk + ln0 + j * 16 + lm;
      const float bv = bias[col];
#pragma unroll
      for (int r = 0; r < 8; ++r) {
        const int row = mblk + lm0 + i * 16 + r + half * 8;
        C[(size_t)row * N + col] = c[i][j][r] + bv;
      }
    }
}

// ---------------- RMS(mult-variant) + RoPE + pack to bf16 ----------------
// grid = (SEQ, NHQ + 2*NKV), block = 32 (one wave); lane owns 4 consecutive d.
__global__ __launch_bounds__(32)
void k_normrope(const float* __restrict__ Qf, const float* __restrict__ Kf,
                const float* __restrict__ Vf, const float* __restrict__ cosT,
                const float* __restrict__ sinT, const float* __restrict__ qw,
                const float* __restrict__ kw, __bf16* __restrict__ Qb,
                __bf16* __restrict__ Kb, __bf16* __restrict__ Vt) {
  const int s = blockIdx.x, y = blockIdx.y, lane = threadIdx.x;
  const int d0 = lane * 4;

  const float* in;
  const float* w = nullptr;
  __bf16* outrow = nullptr;
  bool isV = false;
  int kvh = 0;
  if (y < NHQ) {
    in = Qf + (size_t)s * (NHQ * HDM) + y * HDM;
    w = qw;
    outrow = Qb + ((size_t)y * SEQ + s) * HDM;
  } else if (y < NHQ + NKV) {
    kvh = y - NHQ;
    in = Kf + (size_t)s * (NKV * HDM) + kvh * HDM;
    w = kw;
    outrow = Kb + ((size_t)kvh * SEQ + s) * HDM;
  } else {
    kvh = y - NHQ - NKV;
    in = Vf + (size_t)s * (NKV * HDM) + kvh * HDM;
    isV = true;
  }

  float x[4];
#pragma unroll
  for (int j = 0; j < 4; ++j) x[j] = in[d0 + j];

  if (isV) {
#pragma unroll
    for (int j = 0; j < 4; ++j)
      Vt[((size_t)kvh * HDM + d0 + j) * SEQ + s] = (__bf16)x[j];
    return;
  }

  float ss = x[0] * x[0] + x[1] * x[1] + x[2] * x[2] + x[3] * x[3];
  ss += __shfl_xor(ss, 16);
  ss += __shfl_xor(ss, 8);
  ss += __shfl_xor(ss, 4);
  ss += __shfl_xor(ss, 2);
  ss += __shfl_xor(ss, 1);
  const float rms = sqrtf(ss * (1.f / HDM) + 1e-6f);  // reference MULTIPLIES by this

  float xn[4];
#pragma unroll
  for (int j = 0; j < 4; ++j) xn[j] = w[d0 + j] * (x[j] * rms);

  const float* cr = cosT + (size_t)s * HDM;
  const float* sr = sinT + (size_t)s * HDM;
#pragma unroll
  for (int j = 0; j < 4; ++j) {
    float other = __shfl_xor(xn[j], 16);            // partner d +/- 64
    float rot = (lane < 16) ? -other : other;       // (-x2, x1)
    float o = xn[j] * cr[d0 + j] + rot * sr[d0 + j];
    outrow[d0 + j] = (__bf16)o;
  }
}

// ---------------- flash attention (causal, GQA) ----------------
// One wave per (head, 16-row q tile). 8 waves per block; grid = NHQ*(SEQ/16)/8.
__global__ __launch_bounds__(256)
void k_attn(const __bf16* __restrict__ Qb, const __bf16* __restrict__ Kb,
            const __bf16* __restrict__ Vt, __bf16* __restrict__ Ob) {
  __shared__ __bf16 plds[8 * 16 * 32];
  const int lane = threadIdx.x & 31, wave = threadIdx.x >> 5;
  const int task = blockIdx.x * 8 + wave;
  const int h = task >> 7;          // SEQ/16 = 128 q-tiles per head
  const int qt = task & 127;
  const int kvh = h >> 2;           // GRP = 4
  const int lm = lane & 15, half = lane >> 4;
  __bf16* P = plds + wave * 16 * 32;

  v16bf qf[4];
#pragma unroll
  for (int t = 0; t < 4; ++t)
    qf[t] = load_a_frag(Qb + ((size_t)h * SEQ + qt * 16 + lm) * HDM + t * 32 + half * 16);

  v8f o[8];
  float mi[8], li[8];
#pragma unroll
  for (int r = 0; r < 8; ++r) { mi[r] = -__builtin_inff(); li[r] = 0.f; }
#pragma unroll
  for (int j = 0; j < 8; ++j) o[j] = (v8f)(0.f);

  const float scale = 11.313708498984760f;  // sqrt(128): scores / HD^-0.5
  const int nkb = (qt * 16 + 15) / 32 + 1;

  for (int kb = 0; kb < nkb; ++kb) {
    const int key0 = kb * 32 + lm;  // this lane's key column (sub-tile 0)
    v8f s0 = (v8f)(0.f), s1 = (v8f)(0.f);
#pragma unroll
    for (int t = 0; t < 4; ++t) {
      v16bf b0 = load_b_frag(Kb + ((size_t)kvh * SEQ + key0) * HDM + t * 32 + half * 16);
      v16bf b1 = load_b_frag(Kb + ((size_t)kvh * SEQ + key0 + 16) * HDM + t * 32 + half * 16);
      s0 = wmma_bf16(qf[t], b0, s0);
      s1 = wmma_bf16(qf[t], b1, s1);
    }
    // online softmax over the 32 keys of this block
#pragma unroll
    for (int r = 0; r < 8; ++r) {
      const int qrow = qt * 16 + r + half * 8;
      float v0 = (key0 <= qrow) ? s0[r] * scale : -__builtin_inff();
      float v1 = (key0 + 16 <= qrow) ? s1[r] * scale : -__builtin_inff();
      float rm = fmaxf(v0, v1);
      rm = fmaxf(rm, __shfl_xor(rm, 1));
      rm = fmaxf(rm, __shfl_xor(rm, 2));
      rm = fmaxf(rm, __shfl_xor(rm, 4));
      rm = fmaxf(rm, __shfl_xor(rm, 8));   // stays within 16-lane half on wave32
      const float mnew = fmaxf(mi[r], rm);
      const float alpha = __expf(mi[r] - mnew);
      const float p0 = __expf(v0 - mnew);
      const float p1 = __expf(v1 - mnew);
      float rs = p0 + p1;
      rs += __shfl_xor(rs, 1);
      rs += __shfl_xor(rs, 2);
      rs += __shfl_xor(rs, 4);
      rs += __shfl_xor(rs, 8);
      li[r] = li[r] * alpha + rs;
      mi[r] = mnew;
#pragma unroll
      for (int j = 0; j < 8; ++j) o[j][r] *= alpha;
      P[(r + half * 8) * 32 + lm] = (__bf16)p0;
      P[(r + half * 8) * 32 + lm + 16] = (__bf16)p1;
    }
    asm volatile("s_wait_dscnt 0x0" ::: "memory");  // LDS stores visible to our re-load
    v16bf pf = load_a_frag(P + lm * 32 + half * 16);
#pragma unroll
    for (int j = 0; j < 8; ++j) {
      v16bf vf = load_b_frag(Vt + ((size_t)kvh * HDM + j * 16 + lm) * SEQ + kb * 32 + half * 16);
      o[j] = wmma_bf16(pf, vf, o[j]);
    }
  }

#pragma unroll
  for (int j = 0; j < 8; ++j)
#pragma unroll
    for (int r = 0; r < 8; ++r) {
      const float val = o[j][r] / li[r];
      Ob[(size_t)(qt * 16 + r + half * 8) * HID + h * HDM + j * 16 + lm] = (__bf16)val;
    }
}

// ---------------- host launch ----------------
extern "C" void kernel_launch(void* const* d_in, const int* in_sizes, int n_in,
                              void* d_out, int out_size, void* d_ws, size_t ws_size,
                              hipStream_t stream) {
  const float* x   = (const float*)d_in[0];
  const float* cosT= (const float*)d_in[1];
  const float* sinT= (const float*)d_in[2];
  // d_in[3] = attention_mask (causal, hardcoded)
  const float* wq  = (const float*)d_in[4];
  const float* bq  = (const float*)d_in[5];
  const float* wk  = (const float*)d_in[6];
  const float* bk  = (const float*)d_in[7];
  const float* wv  = (const float*)d_in[8];
  const float* bv  = (const float*)d_in[9];
  const float* wo  = (const float*)d_in[10];
  const float* bo  = (const float*)d_in[11];
  const float* qnw = (const float*)d_in[12];
  const float* knw = (const float*)d_in[13];
  float* out = (float*)d_out;

  char* wsp = (char*)d_ws;
  auto alloc = [&](size_t bytes) -> void* {
    void* p = (void*)wsp;
    wsp += (bytes + 255) & ~(size_t)255;
    return p;
  };
  __bf16* xb  = (__bf16*)alloc((size_t)SEQ * HID * 2);
  __bf16* wqT = (__bf16*)alloc((size_t)HID * (NHQ * HDM) * 2);
  __bf16* wkT = (__bf16*)alloc((size_t)HID * (NKV * HDM) * 2);
  __bf16* wvT = (__bf16*)alloc((size_t)HID * (NKV * HDM) * 2);
  __bf16* woT = (__bf16*)alloc((size_t)(NHQ * HDM) * HID * 2);
  float*  Qf  = (float*)alloc((size_t)SEQ * (NHQ * HDM) * 4);
  float*  Kf  = (float*)alloc((size_t)SEQ * (NKV * HDM) * 4);
  float*  Vf  = (float*)alloc((size_t)SEQ * (NKV * HDM) * 4);
  __bf16* Qb  = (__bf16*)alloc((size_t)NHQ * SEQ * HDM * 2);
  __bf16* Kb  = (__bf16*)alloc((size_t)NKV * SEQ * HDM * 2);
  __bf16* Vt  = (__bf16*)alloc((size_t)NKV * HDM * SEQ * 2);
  __bf16* Ob  = (__bf16*)alloc((size_t)SEQ * HID * 2);

  // 1) conversions
  {
    int n = SEQ * HID;
    k_f32_to_bf16<<<(n + 255) / 256, 256, 0, stream>>>(x, xb, n);
  }
  k_transpose_bf16<<<((long)HID * (NHQ * HDM) + 255) / 256, 256, 0, stream>>>(wq, wqT, HID, NHQ * HDM);
  k_transpose_bf16<<<((long)HID * (NKV * HDM) + 255) / 256, 256, 0, stream>>>(wk, wkT, HID, NKV * HDM);
  k_transpose_bf16<<<((long)HID * (NKV * HDM) + 255) / 256, 256, 0, stream>>>(wv, wvT, HID, NKV * HDM);
  k_transpose_bf16<<<((long)(NHQ * HDM) * HID + 255) / 256, 256, 0, stream>>>(wo, woT, NHQ * HDM, HID);

  // 2) QKV projections (WMMA GEMM, async-LDS staged)
  k_gemm_bf16<<<dim3((NHQ * HDM) / 256, SEQ / 128), 256, 0, stream>>>(xb, wqT, bq, Qf, SEQ, NHQ * HDM, HID);
  k_gemm_bf16<<<dim3((NKV * HDM) / 256, SEQ / 128), 256, 0, stream>>>(xb, wkT, bk, Kf, SEQ, NKV * HDM, HID);
  k_gemm_bf16<<<dim3((NKV * HDM) / 256, SEQ / 128), 256, 0, stream>>>(xb, wvT, bv, Vf, SEQ, NKV * HDM, HID);

  // 3) RMS(mult) + RoPE + pack
  k_normrope<<<dim3(SEQ, NHQ + 2 * NKV), 32, 0, stream>>>(Qf, Kf, Vf, cosT, sinT, qnw, knw, Qb, Kb, Vt);

  // 4) causal GQA flash attention (WMMA)
  k_attn<<<(NHQ * (SEQ / 16)) / 8, 256, 0, stream>>>(Qb, Kb, Vt, Ob);

  // 5) output projection (WMMA GEMM) -> d_out
  k_gemm_bf16<<<dim3(HID / 256, SEQ / 128), 256, 0, stream>>>(Ob, woT, bo, out, SEQ, HID, HID);

  (void)in_sizes; (void)n_in; (void)out_size; (void)ws_size;
}
